// RobustGCN_19911468384631
// MI455X (gfx1250) — matrix operand order repaired and checked by reading, compiled, not run on hardware
//
#include <hip/hip_runtime.h>
#include <hip/hip_bf16.h>
#include <math.h>

#define NN   100000
#define FIN  512
#define HID  256
#define CLS  64

typedef __attribute__((ext_vector_type(16))) __bf16        v16bf;
typedef __attribute__((ext_vector_type(8)))  float         v8f;
typedef __attribute__((ext_vector_type(4)))  unsigned int  u32x4;

union Frag16 {
    v16bf bf;
    u32x4 q[2];
};

__device__ __forceinline__ unsigned short f2bf(float f) {
    unsigned int u = __float_as_uint(f);
    u = (u + 0x7FFFu + ((u >> 16) & 1u)) >> 16;   // round-to-nearest-even
    return (unsigned short)u;
}

// ---------------------------------------------------------------- utilities
__global__ __launch_bounds__(256) void cvt_bf16_kernel(const float* __restrict__ s,
                                                       unsigned short* __restrict__ d, int n) {
    int i = blockIdx.x * 256 + threadIdx.x;
    if (i < n) d[i] = f2bf(s[i]);
}

__global__ __launch_bounds__(256) void fill_f32_kernel(float* __restrict__ p, float v, int n) {
    int i = blockIdx.x * 256 + threadIdx.x;
    if (i < n) p[i] = v;
}

__global__ __launch_bounds__(256) void degree_kernel(const int* __restrict__ src,
                                                     float* __restrict__ deg, int E) {
    int e = blockIdx.x * 256 + threadIdx.x;
    if (e < E) atomicAdd(&deg[src[e]], 1.0f);
}

__global__ __launch_bounds__(256) void invdeg_kernel(const float* __restrict__ deg,
                                                     float* __restrict__ dis,
                                                     float* __restrict__ di, int n) {
    int i = blockIdx.x * 256 + threadIdx.x;
    if (i < n) {
        float d = deg[i];               // >= 1 always (self loop)
        dis[i] = rsqrtf(d);
        di[i]  = 1.0f / d;
    }
}

// ------------------------------------------------- layer 0: dual GEMM (bf16 WMMA)
// mean0 = elu(x@Wm0+bm0), var0 = relu(x@Wv0+bv0); bf16 outputs [NN x HID]
// block: 8 waves, tile = 32 rows x 256 cols; wave = 16 rows x 64 cols, both W's.
// Double-buffered LDS x-tile (one barrier per K-step); all 8 B fragments
// preloaded per K-step so WMMAs overlap the load clause via partial waits.
__global__ __launch_bounds__(256) void gemm0_kernel(
    const float* __restrict__ x,
    const unsigned short* __restrict__ Wm, const unsigned short* __restrict__ Wv,
    const float* __restrict__ bm, const float* __restrict__ bv,
    unsigned short* __restrict__ outM, unsigned short* __restrict__ outV)
{
    __shared__ unsigned short lx[2][32 * 40];   // 2 x (32 rows x 32 k, padded to 40 => 80B rows)

    const int tid  = threadIdx.x;
    const int lane = tid & 31;
    const int wave = tid >> 5;
    const int mB   = blockIdx.x * 32;
    const int m0   = mB + (wave >> 2) * 16;
    const int n0   = (wave & 3) * 64;

    v8f accM[4] = {};
    v8f accV[4] = {};

    const int lr = tid >> 3;                    // staging: row 0..31
    const int lc = (tid & 7) * 4;               // staging: k 0,4,...,28
    const float* xp = x + (size_t)(mB + lr) * FIN + lc;

    const int rowA = (wave >> 2) * 16 + (lane & 15);
    const int kb   = (lane >> 4) * 8;           // A layout: lanes 16-31 hold K+8

    // prologue: stage tile 0
    {
        const float4 v = *(const float4*)(xp);
        unsigned short* p = &lx[0][lr * 40 + lc];
        p[0] = f2bf(v.x); p[1] = f2bf(v.y); p[2] = f2bf(v.z); p[3] = f2bf(v.w);
    }
    __syncthreads();

    const int NIT = FIN / 32;
    for (int it = 0; it < NIT; ++it) {
        const int kk = it * 32;

        // stage next tile into the other buffer (overlaps with WMMAs below)
        if (it + 1 < NIT) {
            if (it + 2 < NIT)
                __builtin_prefetch(xp + kk + 64, 0, 0);
            const float4 v = *(const float4*)(xp + kk + 32);
            unsigned short* p = &lx[(it + 1) & 1][lr * 40 + lc];
            p[0] = f2bf(v.x); p[1] = f2bf(v.y); p[2] = f2bf(v.z); p[3] = f2bf(v.w);
        }

        // preload all 8 B fragments for this K-step
        Frag16 bm8[4], bv8[4];
        #pragma unroll
        for (int nt = 0; nt < 4; ++nt) {
            const unsigned short* pm = Wm + (size_t)(kk + kb) * HID + n0 + nt * 16 + (lane & 15);  // N = lane&15, K in table order
            const unsigned short* pv = Wv + (size_t)(kk + kb) * HID + n0 + nt * 16 + (lane & 15);
            #pragma unroll
            for (int j = 0; j < 4; ++j) {
                bm8[nt].q[0][j] = (unsigned)pm[(2 * j) * HID] | ((unsigned)pm[(2 * j + 1) * HID] << 16);
                bm8[nt].q[1][j] = (unsigned)pm[(16 + 2 * j) * HID] | ((unsigned)pm[(17 + 2 * j) * HID] << 16);
                bv8[nt].q[0][j] = (unsigned)pv[(2 * j) * HID] | ((unsigned)pv[(2 * j + 1) * HID] << 16);
                bv8[nt].q[1][j] = (unsigned)pv[(16 + 2 * j) * HID] | ((unsigned)pv[(17 + 2 * j) * HID] << 16);
            }
        }

        // A fragment from current LDS buffer
        Frag16 a;
        const unsigned short* ap = &lx[it & 1][rowA * 40 + kb];
        a.q[0] = *(const u32x4*)ap;            // K {kb..kb+7}
        a.q[1] = *(const u32x4*)(ap + 16);     // K {kb+16..kb+23}

        #pragma unroll
        for (int nt = 0; nt < 4; ++nt) {
            accM[nt] = __builtin_amdgcn_wmma_f32_16x16x32_bf16(false, a.bf, false, bm8[nt].bf,
                                                               (short)0, accM[nt], false, false);
            accV[nt] = __builtin_amdgcn_wmma_f32_16x16x32_bf16(false, a.bf, false, bv8[nt].bf,
                                                               (short)0, accV[nt], false, false);
        }
        __syncthreads();
    }

    // epilogue: C layout -> lanes 0-15: M=r, lanes 16-31: M=r+8; N = (lane&15)
    const int rOff = (lane >> 4) * 8;
    const int colL = lane & 15;
    #pragma unroll
    for (int nt = 0; nt < 4; ++nt) {
        const int col = n0 + nt * 16 + colL;
        const float bmv = bm[col], bvv = bv[col];
        #pragma unroll
        for (int r = 0; r < 8; ++r) {
            const size_t row = (size_t)(m0 + rOff + r);
            float m = accM[nt][r] + bmv;
            m = (m > 0.0f) ? m : expm1f(m);             // elu
            float v = accV[nt][r] + bvv;
            v = (v > 0.0f) ? v : 0.0f;                  // relu
            outM[row * HID + col] = f2bf(m);
            outV[row * HID + col] = f2bf(v);
        }
    }
}

// ------------------------------------------------- layer 1: dual GEMM + attn fuse
// mean1 = elu(mean0@Wm1+bm1); var1 = relu(var0@Wv1+bv1)+1e-6; attn=exp(-var1)
// store mpre = mean1*attn, vpre = var1*attn^2  (fp32, [NN x CLS])
__global__ __launch_bounds__(256) void gemm1_kernel(
    const unsigned short* __restrict__ m0b, const unsigned short* __restrict__ v0b,
    const unsigned short* __restrict__ Wm, const unsigned short* __restrict__ Wv,
    const float* __restrict__ bm, const float* __restrict__ bv,
    float* __restrict__ mpre, float* __restrict__ vpre)
{
    const int lane  = threadIdx.x & 31;
    const int strip = blockIdx.x * 8 + (threadIdx.x >> 5);
    const int m0    = strip * 16;
    if (m0 >= NN) return;

    v8f accM[4] = {};
    v8f accV[4] = {};

    const size_t rowA = (size_t)(m0 + (lane & 15)) * HID;
    const int    kb   = (lane >> 4) * 8;

    for (int kk = 0; kk < HID; kk += 32) {
        // preload both A fragments and all 8 B fragments before the WMMA burst
        Frag16 am, av;
        am.q[0] = *(const u32x4*)(m0b + rowA + kk + kb);
        am.q[1] = *(const u32x4*)(m0b + rowA + kk + kb + 16);
        av.q[0] = *(const u32x4*)(v0b + rowA + kk + kb);
        av.q[1] = *(const u32x4*)(v0b + rowA + kk + kb + 16);

        Frag16 b0[4], b1[4];
        #pragma unroll
        for (int nt = 0; nt < 4; ++nt) {
            const unsigned short* pm = Wm + (size_t)(kk + kb) * CLS + nt * 16 + (lane & 15);
            const unsigned short* pv = Wv + (size_t)(kk + kb) * CLS + nt * 16 + (lane & 15);
            #pragma unroll
            for (int j = 0; j < 4; ++j) {
                b0[nt].q[0][j] = (unsigned)pm[(2 * j) * CLS] | ((unsigned)pm[(2 * j + 1) * CLS] << 16);
                b0[nt].q[1][j] = (unsigned)pm[(16 + 2 * j) * CLS] | ((unsigned)pm[(17 + 2 * j) * CLS] << 16);
                b1[nt].q[0][j] = (unsigned)pv[(2 * j) * CLS] | ((unsigned)pv[(2 * j + 1) * CLS] << 16);
                b1[nt].q[1][j] = (unsigned)pv[(16 + 2 * j) * CLS] | ((unsigned)pv[(17 + 2 * j) * CLS] << 16);
            }
        }

        #pragma unroll
        for (int nt = 0; nt < 4; ++nt) {
            accM[nt] = __builtin_amdgcn_wmma_f32_16x16x32_bf16(false, am.bf, false, b0[nt].bf,
                                                               (short)0, accM[nt], false, false);
            accV[nt] = __builtin_amdgcn_wmma_f32_16x16x32_bf16(false, av.bf, false, b1[nt].bf,
                                                               (short)0, accV[nt], false, false);
        }
    }

    const int rOff = (lane >> 4) * 8;
    const int colL = lane & 15;
    #pragma unroll
    for (int nt = 0; nt < 4; ++nt) {
        const int col = nt * 16 + colL;
        const float bmv = bm[col], bvv = bv[col];
        #pragma unroll
        for (int r = 0; r < 8; ++r) {
            const size_t row = (size_t)(m0 + rOff + r);
            float m = accM[nt][r] + bmv;
            m = (m > 0.0f) ? m : expm1f(m);
            float v = accV[nt][r] + bvv;
            v = ((v > 0.0f) ? v : 0.0f) + 1e-6f;
            const float a = __expf(-v);
            mpre[row * CLS + col] = m * a;
            vpre[row * CLS + col] = v * a * a;
        }
    }
}

// ------------------------------------------------- edge scatter (dual SpMM)
__global__ __launch_bounds__(256) void spmm_kernel(
    const int* __restrict__ esrc, const int* __restrict__ edst,
    const float* __restrict__ dis, const float* __restrict__ di,
    const float* __restrict__ mpre, const float* __restrict__ vpre,
    float* __restrict__ magg, float* __restrict__ vagg, int E)
{
    const int lane = threadIdx.x & 31;
    const int wid  = (blockIdx.x * 256 + threadIdx.x) >> 5;
    const int nw   = (gridDim.x * 256) >> 5;
    for (int e = wid; e < E; e += nw) {
        const int s = esrc[e], d = edst[e];
        const float w0 = dis[s] * dis[d];
        const float w1 = di[s]  * di[d];
        const size_t sb = (size_t)s * CLS, db = (size_t)d * CLS;
        atomicAdd(&magg[sb + lane],      w0 * mpre[db + lane]);
        atomicAdd(&magg[sb + lane + 32], w0 * mpre[db + lane + 32]);
        atomicAdd(&vagg[sb + lane],      w1 * vpre[db + lane]);
        atomicAdd(&vagg[sb + lane + 32], w1 * vpre[db + lane + 32]);
    }
}

// ------------------------------------------------- epilogue: self-loop + sample + log_softmax
__global__ __launch_bounds__(256) void final_kernel(
    const float* __restrict__ magg, const float* __restrict__ vagg,
    const float* __restrict__ mpre, const float* __restrict__ vpre,
    const float* __restrict__ dis, const float* __restrict__ di,
    const float* __restrict__ sample, float* __restrict__ out)
{
    const int lane = threadIdx.x & 31;
    const int i    = blockIdx.x * 8 + (threadIdx.x >> 5);
    if (i >= NN) return;
    const float sw0 = dis[i] * dis[i];        // self-loop weight D^-0.5 * D^-0.5
    const float sw1 = di[i]  * di[i];
    const size_t b  = (size_t)i * CLS;
    float o[2];
    #pragma unroll
    for (int j = 0; j < 2; ++j) {
        const int c = lane + j * 32;
        const float mt = magg[b + c] + sw0 * mpre[b + c];
        const float vt = vagg[b + c] + sw1 * vpre[b + c];
        o[j] = mt + sample[b + c] * sqrtf(vt);
    }
    float mx = fmaxf(o[0], o[1]);
    #pragma unroll
    for (int off = 16; off > 0; off >>= 1) mx = fmaxf(mx, __shfl_xor(mx, off, 32));
    float s = expf(o[0] - mx) + expf(o[1] - mx);
    #pragma unroll
    for (int off = 16; off > 0; off >>= 1) s += __shfl_xor(s, off, 32);
    const float l = mx + logf(s);
    out[b + lane]      = o[0] - l;
    out[b + lane + 32] = o[1] - l;
}

// ---------------------------------------------------------------- launcher
extern "C" void kernel_launch(void* const* d_in, const int* in_sizes, int n_in,
                              void* d_out, int out_size, void* d_ws, size_t ws_size,
                              hipStream_t stream)
{
    const float* x      = (const float*)d_in[0];
    const float* Wm0    = (const float*)d_in[1];
    const float* bm0    = (const float*)d_in[2];
    const float* Wv0    = (const float*)d_in[3];
    const float* bv0    = (const float*)d_in[4];
    const float* Wm1    = (const float*)d_in[5];
    const float* bm1    = (const float*)d_in[6];
    const float* Wv1    = (const float*)d_in[7];
    const float* bv1    = (const float*)d_in[8];
    const float* sample = (const float*)d_in[9];
    const int*   esrc   = (const int*)d_in[10];
    const int*   edst   = (const int*)d_in[11];
    const int    E      = in_sizes[10];
    float*       out    = (float*)d_out;

    // workspace carve-out (~207 MB)
    char*  ws  = (char*)d_ws;
    size_t off = 0;
    auto take = [&](size_t bytes) -> char* {
        char* p = ws + off;
        off = (off + bytes + 255) & ~(size_t)255;
        return p;
    };
    unsigned short* Wm0b = (unsigned short*)take((size_t)FIN * HID * 2);
    unsigned short* Wv0b = (unsigned short*)take((size_t)FIN * HID * 2);
    unsigned short* Wm1b = (unsigned short*)take((size_t)HID * CLS * 2);
    unsigned short* Wv1b = (unsigned short*)take((size_t)HID * CLS * 2);
    float*          deg  = (float*)take((size_t)NN * 4);
    float*          dis  = (float*)take((size_t)NN * 4);
    float*          di   = (float*)take((size_t)NN * 4);
    unsigned short* m0b  = (unsigned short*)take((size_t)NN * HID * 2);
    unsigned short* v0b  = (unsigned short*)take((size_t)NN * HID * 2);
    float*          mpre = (float*)take((size_t)NN * CLS * 4);
    float*          vpre = (float*)take((size_t)NN * CLS * 4);
    float*          magg = (float*)take((size_t)NN * CLS * 4);
    float*          vagg = (float*)take((size_t)NN * CLS * 4);

    // 1) weights -> bf16
    cvt_bf16_kernel<<<(FIN * HID + 255) / 256, 256, 0, stream>>>(Wm0, Wm0b, FIN * HID);
    cvt_bf16_kernel<<<(FIN * HID + 255) / 256, 256, 0, stream>>>(Wv0, Wv0b, FIN * HID);
    cvt_bf16_kernel<<<(HID * CLS + 255) / 256, 256, 0, stream>>>(Wm1, Wm1b, HID * CLS);
    cvt_bf16_kernel<<<(HID * CLS + 255) / 256, 256, 0, stream>>>(Wv1, Wv1b, HID * CLS);

    // 2) degree (self loop -> init 1.0), dis = deg^-0.5, di = deg^-1
    fill_f32_kernel<<<(NN + 255) / 256, 256, 0, stream>>>(deg, 1.0f, NN);
    degree_kernel<<<(E + 255) / 256, 256, 0, stream>>>(esrc, deg, E);
    invdeg_kernel<<<(NN + 255) / 256, 256, 0, stream>>>(deg, dis, di, NN);

    // 3) zero aggregation buffers (poisoned by harness)
    fill_f32_kernel<<<((NN * CLS) + 255) / 256, 256, 0, stream>>>(magg, 0.0f, NN * CLS);
    fill_f32_kernel<<<((NN * CLS) + 255) / 256, 256, 0, stream>>>(vagg, 0.0f, NN * CLS);

    // 4) layer 0 dual GEMM (WMMA bf16)
    gemm0_kernel<<<NN / 32, 256, 0, stream>>>(x, Wm0b, Wv0b, bm0, bv0, m0b, v0b);

    // 5) layer 1 dual GEMM + variance attention fuse (WMMA bf16)
    gemm1_kernel<<<(NN / 16 + 7) / 8, 256, 0, stream>>>(m0b, v0b, Wm1b, Wv1b, bm1, bv1, mpre, vpre);

    // 6) edge scatter (tables live in 192MB L2)
    spmm_kernel<<<12800, 256, 0, stream>>>(esrc, edst, dis, di, mpre, vpre, magg, vagg, E);

    // 7) self loops + sample + log_softmax
    final_kernel<<<(NN + 7) / 8, 256, 0, stream>>>(magg, vagg, mpre, vpre, dis, di, sample, out);
}
